// NeuroSATNetwork_171798692130
// MI455X (gfx1250) — compile-verified, matrix operand order checked by reading, not run on hardware
//
#include <hip/hip_runtime.h>
#include <math.h>

// ---------------- problem constants (match reference) ----------------
#define DD      64
#define TSTEPS  26
#define NGRAPH  32
#define NLITS   32000      // 32 * 2 * 500
#define NCLS    67200      // 32 * 2100
#define NEDGE   201600     // 3 * NCLS
#define PERG    1000       // literals per graph (2*NV)

typedef __attribute__((ext_vector_type(16))) _Float16 v16h;
typedef __attribute__((ext_vector_type(8)))  _Float16 v8h;
typedef __attribute__((ext_vector_type(4)))  _Float16 v4h;
typedef __attribute__((ext_vector_type(8)))  float    v8f;

__device__ __forceinline__ float sigf(float x) { return 1.0f / (1.0f + __expf(-x)); }

// Build a WMMA A/B fragment (16 halves) from LDS, row-major f16 tile.
// s points at &tile[row*K + k0 + kb0]; halves 0..7 = +0..7, halves 8..15 = +16..23.
__device__ __forceinline__ v16h frag_from_lds(const _Float16* s)
{
    const v8h* p = (const v8h*)s;
    v8h lo = p[0];          // ds_load_b128
    v8h hi = p[2];          // ds_load_b128 (+32 bytes)
    return __builtin_shufflevector(lo, hi, 0,1,2,3,4,5,6,7,8,9,10,11,12,13,14,15);
}

// Cooperative stage of a 16-row f32 tile (16 x K) into LDS as f16 (row-major).
__device__ __forceinline__ void stage_tile(const float* __restrict__ Ab,
                                           _Float16* sA, int K, int tid)
{
    for (int idx = tid * 4; idx < 16 * K; idx += 512) {
        const float4 v = *(const float4*)(Ab + idx);
        v4h h;
        h[0] = (_Float16)v.x; h[1] = (_Float16)v.y;
        h[2] = (_Float16)v.z; h[3] = (_Float16)v.w;
        *(v4h*)(sA + idx) = h;                     // ds_store_b64
    }
}

// ============================================================================
// Fully fused LSTM update:
//   gates = [h | (flip) | aggr | h] @ Wcat  (16x256 stripe per block, WMMA)
//   then LSTM cell + LayerNorm, h/c updated directly. No global intermediate.
// Block = 128 threads (4 waves); grid.x = M/16.  KDIM = 192 (C) or 256 (L).
// For KDIM==256, segment 1 reads H[flip[row]] (cross-block) -> caller must
// double-buffer H (h_in read-only, h_out separate).
// ============================================================================
template<int KDIM>
__global__ __launch_bounds__(128)
void gates_lstm_ln(const float* __restrict__ h_in,
                   const float* __restrict__ aggr,
                   const int*   __restrict__ flip,      // used only when KDIM==256
                   const _Float16* __restrict__ Bfrag,  // [KDIM x 256] frag-packed
                   const float* __restrict__ bih, const float* __restrict__ bhh,
                   const float* __restrict__ lng, const float* __restrict__ lnb,
                   float* __restrict__ h_out, float* __restrict__ c_io)
{
    __shared__ _Float16 sA[16 * KDIM];
    __shared__ float    sG[16 * 256];
    const int tid  = threadIdx.x;
    const int wave = tid >> 5;
    const int lane = tid & 31;
    const int tm   = blockIdx.x;
    const long rbase = (long)tm * 16;

    // ---- stage concatenated input tile (f32 sources -> f16 LDS), fused concat ----
    for (int idx = tid * 4; idx < 16 * KDIM; idx += 512) {
        const int r  = idx / KDIM;            // KDIM is compile-time constant
        const int c  = idx - r * KDIM;
        const int seg = c >> 6, sc = c & 63;
        const long R = rbase + r;
        const float* src;
        long row;
        if (KDIM == 192) {                    // [h | aggr | h]
            src = (seg == 1) ? aggr : h_in;
            row = R;
        } else {                              // [h | h[flip] | aggr | h]
            if (seg == 1)      { src = h_in; row = flip[R]; }
            else if (seg == 2) { src = aggr; row = R; }
            else               { src = h_in; row = R; }
        }
        const float4 v = *(const float4*)(src + row * 64 + sc);
        v4h h;
        h[0] = (_Float16)v.x; h[1] = (_Float16)v.y;
        h[2] = (_Float16)v.z; h[3] = (_Float16)v.w;
        *(v4h*)(sA + idx) = h;
    }
    __syncthreads();

    // ---- 16 x 256 gate stripe via WMMA (4 tiles per wave) ----
    const int kb0  = (lane < 16) ? 0 : 8;
    const int rowl = lane & 15;
    v8f acc[4];
#pragma unroll
    for (int j = 0; j < 4; ++j) acc[j] = (v8f){};

    for (int k0 = 0; k0 < KDIM; k0 += 32) {
        __builtin_prefetch(Bfrag + ((long)((k0 >> 5) + 1) * 16) * 512, 0, 0);
        const v16h a = frag_from_lds(&sA[rowl * KDIM + k0 + kb0]);
#pragma unroll
        for (int j = 0; j < 4; ++j) {
            const int tn = wave * 4 + j;
            const v16h b = *(const v16h*)(Bfrag + (((long)(k0 >> 5) * 16 + tn) << 9) + (lane << 4));
            acc[j] = __builtin_amdgcn_wmma_f32_16x16x32_f16(
                         false, a, false, b, (short)0, acc[j], false, false);
        }
    }

    // spill gate stripe to LDS for cross-wave access
    const int m0l = (lane < 16) ? 0 : 8;
#pragma unroll
    for (int j = 0; j < 4; ++j) {
        const int col = (wave * 4 + j) * 16 + rowl;
#pragma unroll
        for (int r = 0; r < 8; ++r)
            sG[(m0l + r) * 256 + col] = acc[j][r];   // ds_store_b32
    }
    __syncthreads();

    // ---- fused LSTM cell + LayerNorm: each wave handles 4 rows ----
#pragma unroll
    for (int rr = 0; rr < 4; ++rr) {
        const int r = wave * 4 + rr;
        const long R = rbase + r;
        const float* g = &sG[r * 256];
        float hn[2];
#pragma unroll
        for (int j = 0; j < 2; ++j) {
            const int d = lane + j * 32;
            float gi = g[d]       + bih[d]       + bhh[d];
            float gf = g[64 + d]  + bih[64 + d]  + bhh[64 + d];
            float gg = g[128 + d] + bih[128 + d] + bhh[128 + d];
            float go = g[192 + d] + bih[192 + d] + bhh[192 + d];
            float c  = c_io[R * 64 + d];
            float c2 = sigf(gf) * c + sigf(gi) * tanhf(gg);
            c_io[R * 64 + d] = c2;
            hn[j] = sigf(go) * tanhf(c2);
        }
        float s  = hn[0] + hn[1];
        float s2 = hn[0] * hn[0] + hn[1] * hn[1];
#pragma unroll
        for (int off = 16; off; off >>= 1) {
            s  += __shfl_xor(s,  off, 32);
            s2 += __shfl_xor(s2, off, 32);
        }
        const float mean = s * (1.0f / DD);
        const float var  = s2 * (1.0f / DD) - mean * mean;
        const float rstd = rsqrtf(var + 1e-5f);
#pragma unroll
        for (int j = 0; j < 2; ++j) {
            const int d = lane + j * 32;
            h_out[R * 64 + d] = (hn[j] - mean) * rstd * lng[d] + lnb[d];
        }
    }
}

// ============================================================================
// Fused 2-layer MLP (64 -> 64 relu -> 64): hidden tile lives in LDS (f16).
// Block = 128 threads, grid.x = M/16.
// ============================================================================
__global__ __launch_bounds__(128)
void mlp_fused(const float* __restrict__ A,
               const _Float16* __restrict__ fB1, const float* __restrict__ b1,
               const _Float16* __restrict__ fB2, const float* __restrict__ b2,
               float* __restrict__ Dst)
{
    __shared__ _Float16 sA[16 * 64];
    __shared__ _Float16 sH[16 * 64];
    const int tid  = threadIdx.x;
    const int wave = tid >> 5;
    const int lane = tid & 31;
    const int tm   = blockIdx.x;

    stage_tile(A + (long)tm * 1024, sA, 64, tid);
    __syncthreads();

    const int kb0  = (lane < 16) ? 0 : 8;
    const int rowl = lane & 15;
    const int col  = wave * 16 + rowl;
    const int m0l  = (lane < 16) ? 0 : 8;

    // ---- layer 1: H = relu(A @ W1 + b1), kept in LDS ----
    v8f acc = {};
#pragma unroll
    for (int k0 = 0; k0 < 64; k0 += 32) {
        const v16h a = frag_from_lds(&sA[rowl * 64 + k0 + kb0]);
        const v16h b = *(const v16h*)(fB1 + (((k0 >> 5) * 4 + wave) << 9) + (lane << 4));
        acc = __builtin_amdgcn_wmma_f32_16x16x32_f16(
                  false, a, false, b, (short)0, acc, false, false);
    }
    const float bv1 = b1[col];
#pragma unroll
    for (int r = 0; r < 8; ++r) {
        float v = acc[r] + bv1;
        if (v < 0.0f) v = 0.0f;
        sH[(m0l + r) * 64 + col] = (_Float16)v;    // ds_store_b16
    }
    __syncthreads();

    // ---- layer 2: Out = H @ W2 + b2 ----
    v8f acc2 = {};
#pragma unroll
    for (int k0 = 0; k0 < 64; k0 += 32) {
        const v16h a = frag_from_lds(&sH[rowl * 64 + k0 + kb0]);
        const v16h b = *(const v16h*)(fB2 + (((k0 >> 5) * 4 + wave) << 9) + (lane << 4));
        acc2 = __builtin_amdgcn_wmma_f32_16x16x32_f16(
                   false, a, false, b, (short)0, acc2, false, false);
    }
    const float bv2 = b2[col];
    float* drow = Dst + ((long)tm * 16 + m0l) * 64 + col;
#pragma unroll
    for (int r = 0; r < 8; ++r)
        drow[(long)r * 64] = acc2[r] + bv2;
}

// ============================================================================
// Single WMMA GEMM with bias+relu (used for the vote hidden layer).
// ============================================================================
__global__ __launch_bounds__(128)
void wmma_gemm1(const float* __restrict__ A, const _Float16* __restrict__ Bfrag,
                const float* __restrict__ bias, float* __restrict__ Dst)
{
    __shared__ _Float16 sA[16 * 64];
    const int tid  = threadIdx.x;
    const int wave = tid >> 5;
    const int lane = tid & 31;
    const int tm   = blockIdx.x;

    stage_tile(A + (long)tm * 1024, sA, 64, tid);
    __syncthreads();

    const int kb0  = (lane < 16) ? 0 : 8;
    const int rowl = lane & 15;
    v8f acc = {};
#pragma unroll
    for (int k0 = 0; k0 < 64; k0 += 32) {
        const v16h a = frag_from_lds(&sA[rowl * 64 + k0 + kb0]);
        const v16h b = *(const v16h*)(Bfrag + (((k0 >> 5) * 4 + wave) << 9) + (lane << 4));
        acc = __builtin_amdgcn_wmma_f32_16x16x32_f16(
                  false, a, false, b, (short)0, acc, false, false);
    }
    const int col = wave * 16 + rowl;
    const int m0l = (lane < 16) ? 0 : 8;
    const float bv = bias[col];
    float* drow = Dst + ((long)tm * 16 + m0l) * 64 + col;
#pragma unroll
    for (int r = 0; r < 8; ++r) {
        float v = acc[r] + bv;
        drow[(long)r * 64] = v > 0.0f ? v : 0.0f;
    }
}

// ============================================================================
// Weight packing into WMMA-fragment order.
//   n = tn*16 + (lane&15);  k = kblock*32 + (lane<16?0:8) + h + (h&8 ? 8 : 0)
// ============================================================================
__global__ void pack_frag(const float* __restrict__ W, _Float16* __restrict__ out,
                          int K, int N)
{
    int t = blockIdx.x * blockDim.x + threadIdx.x;
    if (t >= K * N) return;
    int h    = t & 15;
    int l    = (t >> 4) & 31;
    int tile = t >> 9;
    int nt   = N >> 4;
    int kb   = tile / nt;
    int tn   = tile - kb * nt;
    int n = tn * 16 + (l & 15);
    int k = kb * 32 + ((l < 16) ? 0 : 8) + h + ((h & 8) ? 8 : 0);
    out[t] = (_Float16)W[(long)k * N + n];
}

// Build concatenated LSTM weight (Ktot x 256) = [Wih.T (Kin rows); Whh.T (64 rows)]
__global__ void build_cat(const float* __restrict__ Wih, const float* __restrict__ Whh,
                          float* __restrict__ out, int Kin, int Ktot)
{
    int t = blockIdx.x * blockDim.x + threadIdx.x;
    if (t >= Ktot * 256) return;
    int k = t >> 8, n = t & 255;
    out[t] = (k < Kin) ? Wih[(long)n * Kin + k] : Whh[(long)n * 64 + (k - Kin)];
}

// ---------------- state init: h = init[d] / sqrt(D), c = 0 ----------------
__global__ void init_state(const float* __restrict__ init, float* __restrict__ h,
                           float* __restrict__ c, int nrows)
{
    int t = blockIdx.x * blockDim.x + threadIdx.x;
    if (t >= nrows * DD) return;
    h[t] = init[t & (DD - 1)] * 0.125f;
    c[t] = 0.0f;
}

// ---------------- edge scatter-add (segment_sum) ----------------
__global__ void scatter_edges(const float* __restrict__ src,
                              const int* __restrict__ isrc, const int* __restrict__ idst,
                              float* __restrict__ dst, int ne)
{
    long t = (long)blockIdx.x * blockDim.x + threadIdx.x;
    if (t >= (long)ne * DD) return;
    int e = (int)(t >> 6), d = (int)(t & 63);
    atomicAdd(&dst[(long)idst[e] * DD + d], src[(long)isrc[e] * DD + d]);
}

// ---------------- final vote: dot(hidden, W2)+b2, segment mean into d_out ----------------
__global__ __launch_bounds__(256)
void vote_reduce(const float* __restrict__ hidden, const float* __restrict__ W2,
                 const float* __restrict__ b2, const int* __restrict__ batch_idx,
                 float* __restrict__ out, int nrows)
{
    int wave = threadIdx.x >> 5, lane = threadIdx.x & 31;
    int row = blockIdx.x * 8 + wave;
    if (row >= nrows) return;
    const float* hr = hidden + (long)row * DD;
    float v = hr[lane] * W2[lane] + hr[lane + 32] * W2[lane + 32];
#pragma unroll
    for (int off = 16; off; off >>= 1) v += __shfl_xor(v, off, 32);
    if (lane == 0)
        atomicAdd(&out[batch_idx[row]], (v + b2[0]) * (1.0f / (float)PERG));
}

// ============================================================================
extern "C" void kernel_launch(void* const* d_in, const int* in_sizes, int n_in,
                              void* d_out, int out_size, void* d_ws, size_t ws_size,
                              hipStream_t stream)
{
    (void)in_sizes; (void)n_in; (void)out_size; (void)ws_size;
    // ------- inputs (setup_inputs order) -------
    const int*   edge_src = (const int*)d_in[0];
    const int*   edge_dst = (const int*)d_in[1];
    const int*   batch_idx= (const int*)d_in[2];
    const int*   flip_idx = (const int*)d_in[3];
    const float* L_init   = (const float*)d_in[6];
    const float* C_init   = (const float*)d_in[7];
    const float* LmsgW1   = (const float*)d_in[8];
    const float* Lmsgb1   = (const float*)d_in[9];
    const float* LmsgW2   = (const float*)d_in[10];
    const float* Lmsgb2   = (const float*)d_in[11];
    const float* CmsgW1   = (const float*)d_in[12];
    const float* Cmsgb1   = (const float*)d_in[13];
    const float* CmsgW2   = (const float*)d_in[14];
    const float* Cmsgb2   = (const float*)d_in[15];
    const float* LupWih   = (const float*)d_in[16];
    const float* LupWhh   = (const float*)d_in[17];
    const float* Lupbih   = (const float*)d_in[18];
    const float* Lupbhh   = (const float*)d_in[19];
    const float* CupWih   = (const float*)d_in[20];
    const float* CupWhh   = (const float*)d_in[21];
    const float* Cupbih   = (const float*)d_in[22];
    const float* Cupbhh   = (const float*)d_in[23];
    const float* Lng      = (const float*)d_in[24];
    const float* Lnb      = (const float*)d_in[25];
    const float* Cng      = (const float*)d_in[26];
    const float* Cnb      = (const float*)d_in[27];
    const float* VoteW1   = (const float*)d_in[28];
    const float* Voteb1   = (const float*)d_in[29];
    const float* VoteW2   = (const float*)d_in[30];
    const float* Voteb2   = (const float*)d_in[31];
    float* out = (float*)d_out;

    // ------- workspace layout -------
    size_t off = 0;
    auto allocF = [&](size_t n) {
        float* p = (float*)((char*)d_ws + off);
        off = (off + n * sizeof(float) + 255) & ~(size_t)255;
        return p;
    };
    auto allocH = [&](size_t n) {
        _Float16* p = (_Float16*)((char*)d_ws + off);
        off = (off + n * sizeof(_Float16) + 255) & ~(size_t)255;
        return p;
    };
    float* L_hA   = allocF((size_t)NLITS * DD);   // ping-pong (flip reads cross blocks)
    float* L_hB   = allocF((size_t)NLITS * DD);
    float* L_c    = allocF((size_t)NLITS * DD);
    float* C_h    = allocF((size_t)NCLS  * DD);   // in-place update is block-local
    float* C_c    = allocF((size_t)NCLS  * DD);
    float* Htmp   = allocF((size_t)NLITS * DD);   // vote hidden
    float* Pre    = allocF((size_t)NCLS  * DD);   // L_pre / C_pre messages
    float* C_aggr = allocF((size_t)NCLS  * DD);
    float* L_aggr = allocF((size_t)NLITS * DD);
    float* Wcat32 = allocF((size_t)256 * 256);
    _Float16* fLmsgW1 = allocH(64 * 64);
    _Float16* fLmsgW2 = allocH(64 * 64);
    _Float16* fCmsgW1 = allocH(64 * 64);
    _Float16* fCmsgW2 = allocH(64 * 64);
    _Float16* fVoteW1 = allocH(64 * 64);
    _Float16* fCcat   = allocH(192 * 256);
    _Float16* fLcat   = allocH(256 * 256);

    const int TB = 256;
    // ------- pack weights (each call; deterministic) -------
    pack_frag<<<(64*64 + TB-1)/TB, TB, 0, stream>>>(LmsgW1, fLmsgW1, 64, 64);
    pack_frag<<<(64*64 + TB-1)/TB, TB, 0, stream>>>(LmsgW2, fLmsgW2, 64, 64);
    pack_frag<<<(64*64 + TB-1)/TB, TB, 0, stream>>>(CmsgW1, fCmsgW1, 64, 64);
    pack_frag<<<(64*64 + TB-1)/TB, TB, 0, stream>>>(CmsgW2, fCmsgW2, 64, 64);
    pack_frag<<<(64*64 + TB-1)/TB, TB, 0, stream>>>(VoteW1, fVoteW1, 64, 64);
    build_cat<<<(192*256 + TB-1)/TB, TB, 0, stream>>>(CupWih, CupWhh, Wcat32, 128, 192);
    pack_frag<<<(192*256 + TB-1)/TB, TB, 0, stream>>>(Wcat32, fCcat, 192, 256);
    build_cat<<<(256*256 + TB-1)/TB, TB, 0, stream>>>(LupWih, LupWhh, Wcat32, 192, 256);
    pack_frag<<<(256*256 + TB-1)/TB, TB, 0, stream>>>(Wcat32, fLcat, 256, 256);

    // ------- init states -------
    init_state<<<(NLITS*DD + TB-1)/TB, TB, 0, stream>>>(L_init, L_hA, L_c, NLITS);
    init_state<<<(NCLS *DD + TB-1)/TB, TB, 0, stream>>>(C_init, C_h, C_c, NCLS);

    const int edgeBlocks = (int)(((long)NEDGE * DD + TB - 1) / TB);
    float* L_cur = L_hA;
    float* L_nxt = L_hB;

    // ------- T message-passing rounds -------
    for (int t = 0; t < TSTEPS; ++t) {
        // L -> C messages: Pre = MLP(L_cur)  (fused, hidden in LDS)
        mlp_fused<<<NLITS/16, 128, 0, stream>>>(L_cur, fLmsgW1, Lmsgb1, fLmsgW2, Lmsgb2, Pre);
        hipMemsetAsync(C_aggr, 0, (size_t)NCLS * DD * sizeof(float), stream);
        scatter_edges<<<edgeBlocks, TB, 0, stream>>>(Pre, edge_src, edge_dst, C_aggr, NEDGE);

        // C update: gates GEMM + LSTM + LN fully fused (in-place, block-local rows)
        gates_lstm_ln<192><<<NCLS/16, 128, 0, stream>>>(
            C_h, C_aggr, nullptr, fCcat, Cupbih, Cupbhh, Cng, Cnb, C_h, C_c);

        // C -> L messages: Pre = MLP(C_h)
        mlp_fused<<<NCLS/16, 128, 0, stream>>>(C_h, fCmsgW1, Cmsgb1, fCmsgW2, Cmsgb2, Pre);
        hipMemsetAsync(L_aggr, 0, (size_t)NLITS * DD * sizeof(float), stream);
        scatter_edges<<<edgeBlocks, TB, 0, stream>>>(Pre, edge_dst, edge_src, L_aggr, NEDGE);

        // L update: fused gates+LSTM+LN; flip reads cross blocks -> ping-pong h
        gates_lstm_ln<256><<<NLITS/16, 128, 0, stream>>>(
            L_cur, L_aggr, flip_idx, fLcat, Lupbih, Lupbhh, Lng, Lnb, L_nxt, L_c);
        float* tmp = L_cur; L_cur = L_nxt; L_nxt = tmp;
    }

    // ------- votes + per-graph mean -------
    wmma_gemm1<<<NLITS/16, 128, 0, stream>>>(L_cur, fVoteW1, Voteb1, Htmp);
    hipMemsetAsync(out, 0, (size_t)NGRAPH * sizeof(float), stream);
    vote_reduce<<<NLITS/8, 256, 0, stream>>>(Htmp, VoteW2, Voteb2, batch_idx, out, NLITS);
}
